// CrossAttention_79242146611919
// MI455X (gfx1250) — matrix operand order repaired and checked by reading, not compiled, not run on hardware
//
#include <hip/hip_runtime.h>
#include <hip/hip_bf16.h>

typedef _Float16 h16;
typedef __attribute__((ext_vector_type(16))) _Float16 v16h;
typedef __attribute__((ext_vector_type(8)))  _Float16 v8h;
typedef __attribute__((ext_vector_type(8)))  float    v8f;
typedef __attribute__((ext_vector_type(4)))  float    v4f;
typedef __attribute__((ext_vector_type(4)))  unsigned int u32x4;
typedef __attribute__((ext_vector_type(8)))  int      i32x8;
typedef __attribute__((ext_vector_type(4)))  int      i32x4;

#define HDIM  1024
#define NHEAD 16
#define DHEAD 64
#define SEQ   2048
#define BATCH 4

#if defined(__has_builtin)
#  if __has_builtin(__builtin_amdgcn_tensor_load_to_lds)
#    define USE_TDM 1
#  endif
#endif
#ifndef USE_TDM
#  define USE_TDM 0
#endif

// ---------------------------------------------------------------- WMMA utils

__device__ __forceinline__ v8f wmma_f16(v16h a, v16h b, v8f c) {
  // D = A(16x32 f16) * B(32x16 f16) + C(16x16 f32)
  return __builtin_amdgcn_wmma_f32_16x16x32_f16(
      /*neg_a=*/false, a, /*neg_b=*/false, b,
      /*c_mod=*/(short)0, c, /*reuse_a=*/false, /*reuse_b=*/false);
}

__device__ __forceinline__ v16h pack16(v8h lo, v8h hi) {
  v16h f;
#pragma unroll
  for (int i = 0; i < 8; ++i) { f[i] = lo[i]; f[i + 8] = hi[i]; }
  return f;
}

// A fragment (16x32, f16 src): lane holds row M=lane&15, h=lane>>4.
// elements: K = kbase+8h+{0..7} then kbase+16+8h+{0..7}   (ISA 7.12.2)
__device__ __forceinline__ v16h load_a_f16(const h16* row, int kbase, int h) {
  const h16* p = row + kbase + 8 * h;
  return pack16(*(const v8h*)p, *(const v8h*)(p + 16));
}

// Same A fragment but converting from f32 source on the fly.
__device__ __forceinline__ v16h load_a_f32cvt(const float* row, int kbase, int h) {
  const float* p = row + kbase + 8 * h;
  v4f a0 = *(const v4f*)(p);
  v4f a1 = *(const v4f*)(p + 4);
  v4f b0 = *(const v4f*)(p + 16);
  v4f b1 = *(const v4f*)(p + 20);
  v16h f;
#pragma unroll
  for (int i = 0; i < 4; ++i) {
    f[i]      = (h16)a0[i];
    f[i + 4]  = (h16)a1[i];
    f[i + 8]  = (h16)b0[i];
    f[i + 12] = (h16)b1[i];
  }
  return f;
}

// B fragment (32x16 f16): caller bakes col (lane&15) into `row` pointer;
// lane half h=lane>>4 holds K = kbase+8h+{0..7} then kbase+16+8h+{0..7}.
__device__ __forceinline__ v16h load_b_f16(const h16* row, int kbase, int h) {
  const h16* p = row + kbase + 8 * h;
  return pack16(*(const v8h*)p, *(const v8h*)(p + 16));
}

// ---------------------------------------------------------------- TDM helper

#if USE_TDM
// DMA a 32(keys) x 64(d) f16 tile from global to LDS via the Tensor Data
// Mover.  Hardware pads +16B after every 128B row so the LDS image has a
// 72-half row stride (bank-conflict-friendly for the column gather).
// D# layout per CDNA5 ISA.
__device__ __forceinline__ void tdm_load_v_tile(const h16* gsrc,
                                                unsigned lds_byte_addr) {
  unsigned long long ga = (unsigned long long)(size_t)gsrc;
  u32x4 g0;
  g0[0] = 1u;                                   // count=1 (valid user D#)
  g0[1] = lds_byte_addr;                        // lds_addr
  g0[2] = (unsigned)(ga & 0xFFFFFFFFu);         // global_addr[31:0]
  g0[3] = (unsigned)((ga >> 32) & 0x01FFFFFFu)  // global_addr[56:32]
        | (2u << 30);                           // type = 2 ("image")
  i32x8 g1;
  g1[0] = (int)((1u << 16)      // data_size = 1 -> 2 bytes
              | (1u << 20)      // pad_enable
              | (4u << 22)      // pad_interval: code 4 -> every 32 DWORDs
              | (3u << 25));    // pad_amount:   code 3 -> 4 DWORDs (16B)
  g1[1] = (int)((unsigned)DHEAD << 16);   // tensor_dim0 = 64
  g1[2] = (int)((unsigned)SEQ << 16);     // tensor_dim1 = 2048
  g1[3] = (int)((unsigned)DHEAD << 16);   // tile_dim0 = 64
  g1[4] = 32;                             // tile_dim1 = 32 (tile_dim2 = 0)
  g1[5] = DHEAD;                          // tensor_dim0_stride = 64
  g1[6] = 0;
  g1[7] = 0;
  i32x4 z4 = {0, 0, 0, 0};
#if __clang_major__ >= 23
  i32x8 z8 = {0, 0, 0, 0, 0, 0, 0, 0};
  __builtin_amdgcn_tensor_load_to_lds(g0, g1, z4, z4, z8, 0);
#else
  __builtin_amdgcn_tensor_load_to_lds(g0, g1, z4, z4, 0);
#endif
}
#endif

// ---------------------------------------------------------------- kernels

__global__ __launch_bounds__(256) void cvt_f32_f16_kernel(
    const float* __restrict__ src, h16* __restrict__ dst, int n4) {
  int i = blockIdx.x * blockDim.x + threadIdx.x;
  if (i < n4) {
    v4f v = *(const v4f*)(src + (size_t)i * 4);
#pragma unroll
    for (int j = 0; j < 4; ++j) dst[(size_t)i * 4 + j] = (h16)v[j];
  }
}

// Y = X @ W^T + bias, output written f16 in head-split layout [B,NH,L,DH].
// Block = 4 waves; wave computes a 16x64 tile (A fragment reused x4).
__global__ __launch_bounds__(128) void proj_kernel(
    const float* __restrict__ X,   // [B*SEQ, H] f32
    const h16*  __restrict__ W,    // [H, H] f16 (row n = W[n][:])
    const float* __restrict__ bias,// [H]
    h16* __restrict__ Out) {       // [B, NH, SEQ, DH] f16
  const int lane = threadIdx.x & 31;
  const int wv   = threadIdx.x >> 5;
  const int hh = lane >> 4, ln = lane & 15;
  const int mbase  = blockIdx.x * 64 + wv * 16;
  const int nbase0 = blockIdx.y * 64;

  const float* Arow = X + (size_t)(mbase + ln) * HDIM;

  v8f acc[4];
#pragma unroll
  for (int t = 0; t < 4; ++t)
#pragma unroll
    for (int i = 0; i < 8; ++i) acc[t][i] = 0.0f;

  for (int kb = 0; kb < HDIM; kb += 32) {
    __builtin_prefetch(Arow + kb + 128, 0, 0);
    v16h a = load_a_f32cvt(Arow, kb, hh);
#pragma unroll
    for (int t = 0; t < 4; ++t) {
      const h16* Wrow = W + (size_t)(nbase0 + t * 16 + ln) * HDIM;
      v16h b = load_b_f16(Wrow, kb, hh);
      acc[t] = wmma_f16(a, b, acc[t]);
    }
  }

  const int bIdx  = mbase >> 11;          // / SEQ (2048)
  const int lrow0 = mbase & (SEQ - 1);    // % SEQ
#pragma unroll
  for (int t = 0; t < 4; ++t) {
    const int n = nbase0 + t * 16 + ln;
    const int head = n >> 6, d = n & 63;
    const float bs = bias[n];
    h16* ob = Out + ((size_t)(bIdx * NHEAD + head) * SEQ) * DHEAD;
#pragma unroll
    for (int r = 0; r < 8; ++r) {
      const int lr = lrow0 + r + 8 * hh;
      ob[(size_t)lr * DHEAD + d] = (h16)(acc[t][r] + bs);
    }
  }
}

// Flash attention: one wave per 16 query rows; streams 32 keys/iteration.
// V tiles are DMA'd into LDS by the Tensor Data Mover (double-buffered).
// Writes heads recombined: Out f16 [B, LQ, H].
__global__ __launch_bounds__(128) void attn_kernel(
    const h16* __restrict__ Q, const h16* __restrict__ K,
    const h16* __restrict__ V,  // [B, NH, SEQ, DH] f16
    const int* __restrict__ mask,  // [B, SEQ]
    h16* __restrict__ Out) {
  const int lane = threadIdx.x & 31;
  const int wv   = threadIdx.x >> 5;
  const int hh = lane >> 4, ln = lane & 15;
  const int qbase = blockIdx.x * 64 + wv * 16;
  const int head = blockIdx.y, b = blockIdx.z;

  const size_t hoff = (size_t)(b * NHEAD + head) * SEQ * DHEAD;
  const h16* Qh = Q + hoff;
  const h16* Kh = K + hoff;
  const h16* Vh = V + hoff;
  const int* mk = mask + (size_t)b * SEQ;

  __shared__ h16 ldsP[4][16][32];              // per-wave P tile (16q x 32k)
  __shared__ h16 ldsV[4][2][32][DHEAD + 8];    // per-wave double-buffered V

#if USE_TDM
  // prologue: DMA first V tile into buffer 0
  tdm_load_v_tile(Vh, (unsigned)(size_t)&ldsV[wv][0][0][0]);
#endif

  // Q fragments for full K-dim (DH=64 -> 2 fragments), loaded once.
  const h16* qrow = Qh + (size_t)(qbase + ln) * DHEAD;
  v16h aq0 = load_a_f16(qrow, 0, hh);
  v16h aq1 = load_a_f16(qrow, 32, hh);

  v8f o[4];
  float mrow[8], rowl[8];
#pragma unroll
  for (int t = 0; t < 4; ++t)
#pragma unroll
    for (int i = 0; i < 8; ++i) o[t][i] = 0.0f;
#pragma unroll
  for (int r = 0; r < 8; ++r) { mrow[r] = -1e30f; rowl[r] = 0.0f; }

  for (int kb = 0; kb < SEQ; kb += 32) {
    const int buf = (kb >> 5) & 1;
#if USE_TDM
    // kick DMA for the *next* tile while we chew on this one
    if (kb + 32 < SEQ)
      tdm_load_v_tile(Vh + (size_t)(kb + 32) * DHEAD,
                      (unsigned)(size_t)&ldsV[wv][buf ^ 1][0][0]);
#else
    {
      const h16* vrow = Vh + (size_t)(kb + lane) * DHEAD;
#pragma unroll
      for (int c = 0; c < DHEAD; c += 8)
        *(v8h*)&ldsV[wv][buf][lane][c] = *(const v8h*)(vrow + c);
    }
#endif

    // ---- S = Q * K^T for 32 keys (two 16x16 score tiles)
    v8f s0, s1;
#pragma unroll
    for (int i = 0; i < 8; ++i) { s0[i] = 0.0f; s1[i] = 0.0f; }
    {
      const h16* k0row = Kh + (size_t)(kb + ln) * DHEAD;
      const h16* k1row = Kh + (size_t)(kb + 16 + ln) * DHEAD;
      s0 = wmma_f16(aq0, load_b_f16(k0row, 0,  hh), s0);
      s0 = wmma_f16(aq1, load_b_f16(k0row, 32, hh), s0);
      s1 = wmma_f16(aq0, load_b_f16(k1row, 0,  hh), s1);
      s1 = wmma_f16(aq1, load_b_f16(k1row, 32, hh), s1);
    }

    // ---- scale + mask (mask==1 -> -1e30 sentinel; exp->0 vs real max)
    const int m0 = mk[kb + ln];
    const int m1 = mk[kb + 16 + ln];
    float f0[8], f1[8];
#pragma unroll
    for (int r = 0; r < 8; ++r) {
      f0[r] = m0 ? -1e30f : s0[r] * 0.125f;
      f1[r] = m1 ? -1e30f : s1[r] * 0.125f;
    }

    // ---- online softmax (rows live across 16 lanes of each wave half)
#pragma unroll
    for (int r = 0; r < 8; ++r) {
      float fm = fmaxf(f0[r], f1[r]);
#pragma unroll
      for (int ofs = 8; ofs >= 1; ofs >>= 1)
        fm = fmaxf(fm, __shfl_xor(fm, ofs, 32));
      const float mnew  = fmaxf(mrow[r], fm);
      const float alpha = __expf(mrow[r] - mnew);
      const float p0 = __expf(f0[r] - mnew);
      const float p1 = __expf(f1[r] - mnew);
      float psum = p0 + p1;
#pragma unroll
      for (int ofs = 8; ofs >= 1; ofs >>= 1)
        psum += __shfl_xor(psum, ofs, 32);
      rowl[r] = rowl[r] * alpha + psum;
      mrow[r] = mnew;
#pragma unroll
      for (int t = 0; t < 4; ++t) o[t][r] *= alpha;
      ldsP[wv][r + 8 * hh][ln]      = (h16)p0;   // stash P for transpose
      ldsP[wv][r + 8 * hh][16 + ln] = (h16)p1;
    }

    // ---- P as A fragment (LDS transpose; per-wave DS ops are in-order)
    v16h pa;
    {
      const h16* prow = &ldsP[wv][ln][0];
      pa = pack16(*(const v8h*)(prow + 8 * hh),
                  *(const v8h*)(prow + 16 + 8 * hh));
    }

#if USE_TDM
    // release the tile we are about to read (TENSORcnt is in-order per wave)
    if (kb + 32 < SEQ)
      asm volatile("s_wait_tensorcnt 0x1" ::: "memory");
    else
      asm volatile("s_wait_tensorcnt 0x0" ::: "memory");
#endif

    // ---- O += P * V  (four 16x16 output tiles over d)
#pragma unroll
    for (int t = 0; t < 4; ++t) {
      v16h bv;
#pragma unroll
      for (int i = 0; i < 16; ++i)
        bv[i] = ldsV[wv][buf][(i < 8 ? 8 * hh + i : 16 + 8 * hh + i - 8)][t * 16 + ln];
      o[t] = wmma_f16(pa, bv, o[t]);
    }
  }

  // ---- normalize and write heads recombined: [B, LQ, H] f16
#pragma unroll
  for (int t = 0; t < 4; ++t) {
#pragma unroll
    for (int r = 0; r < 8; ++r) {
      const int row = qbase + r + 8 * hh;
      const float val = o[t][r] / rowl[r];
      Out[((size_t)b * SEQ + row) * HDIM + head * DHEAD + t * 16 + ln] = (h16)val;
    }
  }
}

// y1 = AO @ Wm^T + bm ; x = y_q + y1 ; LayerNorm(x) -> out (f32)
// Block (8 waves) owns 16 full rows: wave wv covers columns [wv*128,+128).
__global__ __launch_bounds__(256) void out_kernel(
    const h16* __restrict__ A,    // [B*LQ, H] f16 (attention output)
    const h16* __restrict__ Wm,   // [H, H] f16
    const float* __restrict__ bm,
    const float* __restrict__ yq, // [B*LQ, H] f32 residual
    const float* __restrict__ g, const float* __restrict__ bt,
    float* __restrict__ out) {
  const int lane = threadIdx.x & 31;
  const int wv   = threadIdx.x >> 5;
  const int hh = lane >> 4, ln = lane & 15;
  const int mbase = blockIdx.x * 16;

  __shared__ float redS[16][8];
  __shared__ float redS2[16][8];

  const h16* Arow = A + (size_t)(mbase + ln) * HDIM;

  v8f acc[8];
#pragma unroll
  for (int t = 0; t < 8; ++t)
#pragma unroll
    for (int i = 0; i < 8; ++i) acc[t][i] = 0.0f;

  for (int kb = 0; kb < HDIM; kb += 32) {
    __builtin_prefetch(Arow + kb + 256, 0, 0);
    v16h a = load_a_f16(Arow, kb, hh);
#pragma unroll
    for (int t = 0; t < 8; ++t) {
      const int n = wv * 128 + t * 16 + ln;
      v16h b = load_b_f16(Wm + (size_t)n * HDIM, kb, hh);
      acc[t] = wmma_f16(a, b, acc[t]);
    }
  }

  // bias + residual, accumulate per-row partial sums for LN
  float ps[8], ps2[8];
#pragma unroll
  for (int r = 0; r < 8; ++r) { ps[r] = 0.0f; ps2[r] = 0.0f; }
#pragma unroll
  for (int t = 0; t < 8; ++t) {
    const int n = wv * 128 + t * 16 + ln;
    const float bs = bm[n];
#pragma unroll
    for (int r = 0; r < 8; ++r) {
      const int row = r + 8 * hh;
      float x = acc[t][r] + bs + yq[(size_t)(mbase + row) * HDIM + n];
      acc[t][r] = x;
      ps[r]  += x;
      ps2[r] += x * x;
    }
  }
  // reduce across the 16 lanes of each half, then across the 8 waves via LDS
#pragma unroll
  for (int r = 0; r < 8; ++r) {
#pragma unroll
    for (int ofs = 8; ofs >= 1; ofs >>= 1) {
      ps[r]  += __shfl_xor(ps[r],  ofs, 32);
      ps2[r] += __shfl_xor(ps2[r], ofs, 32);
    }
  }
  if (ln == 0) {
#pragma unroll
    for (int r = 0; r < 8; ++r) {
      redS [r + 8 * hh][wv] = ps[r];
      redS2[r + 8 * hh][wv] = ps2[r];
    }
  }
  __syncthreads();

  float mu[8], inv[8];
#pragma unroll
  for (int r = 0; r < 8; ++r) {
    const int row = r + 8 * hh;
    float s = 0.0f, s2 = 0.0f;
#pragma unroll
    for (int w = 0; w < 8; ++w) { s += redS[row][w]; s2 += redS2[row][w]; }
    mu[r] = s * (1.0f / HDIM);
    const float var = s2 * (1.0f / HDIM) - mu[r] * mu[r];
    inv[r] = rsqrtf(var + 1e-5f);
  }

#pragma unroll
  for (int t = 0; t < 8; ++t) {
    const int n = wv * 128 + t * 16 + ln;
    const float gg = g[n], bb = bt[n];
#pragma unroll
    for (int r = 0; r < 8; ++r) {
      const int row = r + 8 * hh;
      out[(size_t)(mbase + row) * HDIM + n] = (acc[t][r] - mu[r]) * inv[r] * gg + bb;
    }
  }
}

// ---------------------------------------------------------------- launch

extern "C" void kernel_launch(void* const* d_in, const int* in_sizes, int n_in,
                              void* d_out, int out_size, void* d_ws, size_t ws_size,
                              hipStream_t stream) {
  const float* x_v  = (const float*)d_in[0];
  const float* x_k  = (const float*)d_in[1];
  const float* y_q  = (const float*)d_in[2];
  const int*   mask = (const int*)  d_in[3];
  const float* Wv   = (const float*)d_in[4];
  const float* bv   = (const float*)d_in[5];
  const float* Wk   = (const float*)d_in[6];
  const float* bk   = (const float*)d_in[7];
  const float* Wq   = (const float*)d_in[8];
  const float* bq   = (const float*)d_in[9];
  const float* Wm   = (const float*)d_in[10];
  const float* bm   = (const float*)d_in[11];
  const float* ln_g = (const float*)d_in[12];
  const float* ln_b = (const float*)d_in[13];
  float* out = (float*)d_out;

  h16* ws = (h16*)d_ws;
  const size_t TEN = (size_t)BATCH * SEQ * HDIM;  // 8,388,608 elems
  const size_t WSZ = (size_t)HDIM * HDIM;         // 1,048,576 elems
  h16* Qh  = ws;
  h16* Kh  = ws + TEN;
  h16* Vh  = ws + 2 * TEN;
  h16* AO  = ws + 3 * TEN;
  h16* Wqh = ws + 4 * TEN;
  h16* Wkh = Wqh + WSZ;
  h16* Wvh = Wkh + WSZ;
  h16* Wmh = Wvh + WSZ;

  // weights -> f16
  {
    const int n4 = (int)(WSZ / 4);
    dim3 g((n4 + 255) / 256), b(256);
    cvt_f32_f16_kernel<<<g, b, 0, stream>>>(Wq, Wqh, n4);
    cvt_f32_f16_kernel<<<g, b, 0, stream>>>(Wk, Wkh, n4);
    cvt_f32_f16_kernel<<<g, b, 0, stream>>>(Wv, Wvh, n4);
    cvt_f32_f16_kernel<<<g, b, 0, stream>>>(Wm, Wmh, n4);
  }

  // Q/K/V projections (WMMA GEMM, head-split f16 output)
  {
    dim3 g(BATCH * SEQ / 64, HDIM / 64), b(128);
    proj_kernel<<<g, b, 0, stream>>>(y_q, Wqh, bq, Qh);
    proj_kernel<<<g, b, 0, stream>>>(x_k, Wkh, bk, Kh);
    proj_kernel<<<g, b, 0, stream>>>(x_v, Wvh, bv, Vh);
  }

  // flash attention (TDM-staged V tiles)
  {
    dim3 g(SEQ / 64, NHEAD, BATCH), b(128);
    attn_kernel<<<g, b, 0, stream>>>(Qh, Kh, Vh, mask, AO);
  }

  // output projection + residual + LayerNorm
  {
    dim3 g(BATCH * SEQ / 16), b(256);
    out_kernel<<<g, b, 0, stream>>>(AO, Wmh, bm, y_q, ln_g, ln_b, out);
  }
}
